// SigmaMoE_47974784697230
// MI455X (gfx1250) — compile-verified, matrix operand order checked by reading, not run on hardware
//
#include <hip/hip_runtime.h>
#include <hip/hip_bf16.h>

// ---- problem constants (match reference) ----
#define Bb 8
#define Ss 2048
#define Dd 1024
#define Ee 32
#define Hh 128
#define Rr 30          // routed experts
#define KTOP 4
#define NSEL 6         // KTOP + 2 shared
#define Tt (Bb*Ss)     // 16384 tokens
#define MTILE 32       // token rows per workgroup (2 WMMA m-tiles)

typedef __bf16 v16bf __attribute__((ext_vector_type(16)));
typedef float  v8f   __attribute__((ext_vector_type(8)));
typedef unsigned int u32x4 __attribute__((ext_vector_type(4)));

// TDM descriptor group types (clang-23 6-arg builtin form)
typedef unsigned int tdm_u32x4 __attribute__((ext_vector_type(4)));
typedef int          tdm_i32x8 __attribute__((ext_vector_type(8)));
typedef int          tdm_i32x4 __attribute__((ext_vector_type(4)));

union AFrag { v16bf v; u32x4 q[2]; };

// ---------------------------------------------------------------------------
// 0) zero output region + per-expert counters
// ---------------------------------------------------------------------------
__global__ void zero_init_kernel(float* __restrict__ out, int* __restrict__ counts) {
    size_t stride = (size_t)gridDim.x * blockDim.x;
    size_t gid = (size_t)blockIdx.x * blockDim.x + threadIdx.x;
    const size_t n = (size_t)Tt * Dd;
    for (size_t i = gid; i < n; i += stride) out[i] = 0.0f;
    if (gid < Ee) counts[gid] = 0;
}

// ---------------------------------------------------------------------------
// 1) f32 -> bf16 converts (x direct; keys/values transposed so both GEMMs
//    read B operands as contiguous N-major rows over K)
// ---------------------------------------------------------------------------
__global__ void convert_x_kernel(const float* __restrict__ x, __bf16* __restrict__ xb) {
    int stride = gridDim.x * blockDim.x;
    const int n = Tt * Dd;
    for (int i = blockIdx.x * blockDim.x + threadIdx.x; i < n; i += stride)
        xb[i] = (__bf16)x[i];
}

__global__ void convert_keys_kernel(const float* __restrict__ keys, __bf16* __restrict__ keysT) {
    // keys [E][D][H] -> keysT [E][H][D]
    int stride = gridDim.x * blockDim.x;
    const int n = Ee * Dd * Hh;
    for (int i = blockIdx.x * blockDim.x + threadIdx.x; i < n; i += stride) {
        int e = i / (Dd * Hh);
        int r = i - e * (Dd * Hh);
        int d = r / Hh;
        int h = r - d * Hh;
        keysT[((size_t)e * Hh + h) * Dd + d] = (__bf16)keys[i];
    }
}

__global__ void convert_values_kernel(const float* __restrict__ vals, __bf16* __restrict__ valsT) {
    // values [E][H][D] -> valuesT [E][D][H]
    int stride = gridDim.x * blockDim.x;
    const int n = Ee * Hh * Dd;
    for (int i = blockIdx.x * blockDim.x + threadIdx.x; i < n; i += stride) {
        int e = i / (Hh * Dd);
        int r = i - e * (Hh * Dd);
        int h = r / Dd;
        int d = r - h * Dd;
        valsT[((size_t)e * Dd + d) * Hh + h] = (__bf16)vals[i];
    }
}

// ---------------------------------------------------------------------------
// 2) Router: one wave32 per token; lane = expert. Sigmoid affinity, top-4
//    over routed experts via wave argmax (jax tie-break: lowest index wins),
//    append shared experts {30,31}. Writes sel indices + scatters (token,
//    weight) into per-expert lists.
// ---------------------------------------------------------------------------
__global__ __launch_bounds__(256)
void router_kernel(const float* __restrict__ sel_in,
                   const float* __restrict__ expert_sel,
                   const float* __restrict__ bias,
                   int* __restrict__ sel_out,
                   int* __restrict__ counts,
                   int* __restrict__ list_tok,
                   float* __restrict__ list_w) {
    const int lane = threadIdx.x & 31;
    const int wave = threadIdx.x >> 5;
    const int token = blockIdx.x * 8 + wave;

    const float4* x4 = (const float4*)(sel_in + (size_t)token * Dd);
    const float4* e4 = (const float4*)(expert_sel + (size_t)lane * Dd);
    float acc = 0.0f;
#pragma unroll 4
    for (int d = 0; d < Dd / 4; ++d) {
        float4 a = x4[d];
        float4 b = e4[d];
        acc += a.x * b.x + a.y * b.y + a.z * b.z + a.w * b.w;
    }
    // sigmoid via v_rcp_f32 (no IEEE div chain)
    float aff = __builtin_amdgcn_rcpf(1.0f + __expf(-acc));
    float score = aff + bias[lane];
    float val = (lane < Rr) ? score : -__builtin_inff();

    int   sidx[NSEL];
    float sw[NSEL];
#pragma unroll
    for (int k = 0; k < KTOP; ++k) {
        float v = val;
        int idx = lane;
#pragma unroll
        for (int off = 16; off > 0; off >>= 1) {
            float ov = __shfl_xor(v, off, 32);
            int   oi = __shfl_xor(idx, off, 32);
            if (ov > v || (ov == v && oi < idx)) { v = ov; idx = oi; }
        }
        sidx[k] = idx;                 // wave-uniform argmax
        sw[k]   = __shfl(aff, idx, 32);
        if (lane == idx) val = -__builtin_inff();
    }
    sidx[KTOP]     = Rr;
    sidx[KTOP + 1] = Rr + 1;
    sw[KTOP]       = __shfl(aff, Rr, 32);
    sw[KTOP + 1]   = __shfl(aff, Rr + 1, 32);

    if (lane == 0) {
#pragma unroll
        for (int k = 0; k < NSEL; ++k) {
            sel_out[(size_t)token * NSEL + k] = sidx[k];
            int pos = atomicAdd(&counts[sidx[k]], 1);
            list_tok[(size_t)sidx[k] * Tt + pos] = token;
            list_w  [(size_t)sidx[k] * Tt + pos] = sw[k];
        }
    }
}

// ---------------------------------------------------------------------------
// 3) Grouped expert GEMM with WMMA bf16, 32-row M tiles.
//    X rows are gathered into LDS by the Tensor Data Mover in gather mode
//    (16-bit row indices = token ids), tracked with TENSORcnt.
//    Each wave computes both m-tiles, sharing every B fragment (halves L2
//    weight traffic vs a 16-row tile).
// ---------------------------------------------------------------------------
__global__ __launch_bounds__(256, 2)
void moe_ffn_kernel(const __bf16* __restrict__ xb,        // [T][D]
                    const __bf16* __restrict__ keysT,     // [E][H][D]
                    const __bf16* __restrict__ valuesT,   // [E][D][H]
                    const int*   __restrict__ counts,
                    const int*   __restrict__ list_tok,
                    const float* __restrict__ list_w,
                    float* __restrict__ out) {            // [T][D]
    const int e = blockIdx.x;
    const int mtile = blockIdx.y;
    const int cnt = counts[e];
    if (mtile * MTILE >= cnt) return;

    __shared__ __align__(16) __bf16 x_lds[MTILE * Dd];    // 64 KB
    __shared__ __align__(16) __bf16 h_lds[MTILE * Hh];    // 8 KB
    __shared__ int   tok_lds[MTILE];
    __shared__ float w_lds[MTILE];

    const int tid  = threadIdx.x;
    const int lane = tid & 31;
    const int wave = tid >> 5;
    const int m16  = lane & 15;
    const int hf   = lane >> 4;

    if (tid < MTILE) {
        int row = mtile * MTILE + tid;
        bool ok = row < cnt;
        tok_lds[tid] = ok ? list_tok[(size_t)e * Tt + row] : 0;
        w_lds[tid]   = ok ? list_w  [(size_t)e * Tt + row] : 0.0f;
    }
    __syncthreads();

    // ---- TDM gather: 2 x (16 rows of D bf16) from xb into x_lds ----
    if (wave == 0) {
        unsigned long long ga = (unsigned long long)(uintptr_t)xb;
        unsigned lds_base = (unsigned)(uintptr_t)&x_lds[0];   // flat[31:0] == LDS offset
#pragma unroll
        for (int g = 0; g < 2; ++g) {
            int p[8];
#pragma unroll
            for (int i = 0; i < 8; ++i) {
                int t0 = __builtin_amdgcn_readfirstlane(tok_lds[g * 16 + 2 * i]);
                int t1 = __builtin_amdgcn_readfirstlane(tok_lds[g * 16 + 2 * i + 1]);
                p[i] = (t0 & 0xFFFF) | (t1 << 16);
            }
            // D# group0: count=1, gather_mode=1, 16-bit idx, lds_addr, global_addr, type=2
            tdm_u32x4 g0 = { 0x80000001u,
                             lds_base + (unsigned)g * (16u * Dd * 2u),
                             (unsigned)(ga & 0xFFFFFFFFu),
                             (unsigned)(((ga >> 32) & 0x01FFFFFFu) | 0x80000000u) };
            // D# group1: data_size=2B; tensor_dim0=1024; tensor_dim1=T;
            //            tile_dim0=1024; tile_dim1=16 idx; dim0_stride=1024
            tdm_i32x8 g1 = { (int)0x00010000,
                             (int)((Dd & 0xFFFF) << 16),
                             (int)((Tt & 0xFFFF) << 16),
                             (int)((Dd & 0xFFFF) << 16),
                             16,
                             Dd,
                             0, 0 };
            tdm_i32x4 g2 = { p[0], p[1], p[2], p[3] };    // row indices 0..7
            tdm_i32x4 g3 = { p[4], p[5], p[6], p[7] };    // row indices 8..15
            tdm_i32x8 gx = { 0, 0, 0, 0, 0, 0, 0, 0 };    // unused extension slot
            __builtin_amdgcn_tensor_load_to_lds(g0, g1, g2, g3, gx, 0);
        }
        __builtin_amdgcn_s_wait_tensorcnt(0);
    }
    __syncthreads();

    // ---- stage 1: h[32 x 16] per wave (H cols 16*wave..+15), K over D ----
    v8f acc0 = 0, acc1 = 0;
    const int nH = wave * 16 + m16;                                   // H column
    const u32x4* bK  = (const u32x4*)(keysT + ((size_t)e * Hh + nH) * Dd);
    const u32x4* aX0 = (const u32x4*)x_lds + m16 * (Dd / 8);          // rows 0..15
    const u32x4* aX1 = (const u32x4*)x_lds + (16 + m16) * (Dd / 8);   // rows 16..31
#pragma unroll 4
    for (int kb = 0; kb < Dd / 32; ++kb) {
        AFrag a0, a1, b;
        const int c0 = kb * 4 + hf;         // (kb*32 + hf*8)/8
        const int c1 = kb * 4 + 2 + hf;     // (kb*32 + 16 + hf*8)/8
        b.q[0]  = bK[c0];  b.q[1]  = bK[c1];
        a0.q[0] = aX0[c0]; a0.q[1] = aX0[c1];
        a1.q[0] = aX1[c0]; a1.q[1] = aX1[c1];
        acc0 = __builtin_amdgcn_wmma_f32_16x16x32_bf16(
            false, a0.v, false, b.v, (short)0, acc0, false, false);
        acc1 = __builtin_amdgcn_wmma_f32_16x16x32_bf16(
            false, a1.v, false, b.v, (short)0, acc1, false, false);
    }
    // silu * routing weight -> h_lds (bf16). Row of acc[i] = i + 8*hf (+16 for m1).
#pragma unroll
    for (int i = 0; i < 8; ++i) {
        int r0 = i + 8 * hf;
        float v0 = acc0[i];
        float s0 = v0 * __builtin_amdgcn_rcpf(1.0f + __expf(-v0));
        h_lds[r0 * Hh + nH] = (__bf16)(s0 * w_lds[r0]);
        int r1 = r0 + 16;
        float v1 = acc1[i];
        float s1 = v1 * __builtin_amdgcn_rcpf(1.0f + __expf(-v1));
        h_lds[r1 * Hh + nH] = (__bf16)(s1 * w_lds[r1]);
    }
    __syncthreads();

    // ---- stage 2: out[32 x 1024]; wave handles 8 n-tiles x 2 m-tiles ----
#pragma unroll 1
    for (int j = 0; j < 8; ++j) {
        const int nt = wave * 8 + j;          // n-tile 0..63
        const int nD = nt * 16 + m16;         // output column
        v8f o0 = 0, o1 = 0;
        const u32x4* bV  = (const u32x4*)(valuesT + ((size_t)e * Dd + nD) * Hh);
        const u32x4* aH0 = (const u32x4*)h_lds + m16 * (Hh / 8);
        const u32x4* aH1 = (const u32x4*)h_lds + (16 + m16) * (Hh / 8);
#pragma unroll
        for (int kb = 0; kb < Hh / 32; ++kb) {
            AFrag a0, a1, b;
            const int c0 = kb * 4 + hf;
            const int c1 = kb * 4 + 2 + hf;
            b.q[0]  = bV[c0];  b.q[1]  = bV[c1];
            a0.q[0] = aH0[c0]; a0.q[1] = aH0[c1];
            a1.q[0] = aH1[c0]; a1.q[1] = aH1[c1];
            o0 = __builtin_amdgcn_wmma_f32_16x16x32_bf16(
                false, a0.v, false, b.v, (short)0, o0, false, false);
            o1 = __builtin_amdgcn_wmma_f32_16x16x32_bf16(
                false, a1.v, false, b.v, (short)0, o1, false, false);
        }
#pragma unroll
        for (int i = 0; i < 8; ++i) {
            int r0 = i + 8 * hf;
            int t0 = tok_lds[r0];             // rows beyond cnt add 0.0f to token 0
            __hip_atomic_fetch_add(&out[(size_t)t0 * Dd + nD], o0[i],
                                   __ATOMIC_RELAXED, __HIP_MEMORY_SCOPE_AGENT);
            int t1 = tok_lds[r0 + 16];
            __hip_atomic_fetch_add(&out[(size_t)t1 * Dd + nD], o1[i],
                                   __ATOMIC_RELAXED, __HIP_MEMORY_SCOPE_AGENT);
        }
    }
}

// ---------------------------------------------------------------------------
// launch
// ---------------------------------------------------------------------------
extern "C" void kernel_launch(void* const* d_in, const int* in_sizes, int n_in,
                              void* d_out, int out_size, void* d_ws, size_t ws_size,
                              hipStream_t stream) {
    const float* token_stream    = (const float*)d_in[0];
    const float* selection_input = (const float*)d_in[1];
    const float* keys            = (const float*)d_in[2];
    const float* values          = (const float*)d_in[3];
    const float* expert_sel      = (const float*)d_in[4];
    const float* bias_ffn        = (const float*)d_in[5];

    float* out = (float*)d_out;                            // [T][D] f32
    int*   sel_out = (int*)d_out + (size_t)Tt * Dd;        // [T][6] int32 (tuple tail)

    // workspace layout
    char* ws = (char*)d_ws;
    __bf16* xb      = (__bf16*)ws;  ws += (size_t)Tt * Dd * sizeof(__bf16);        // 32 MB
    __bf16* keysT   = (__bf16*)ws;  ws += (size_t)Ee * Hh * Dd * sizeof(__bf16);   // 8 MB
    __bf16* valuesT = (__bf16*)ws;  ws += (size_t)Ee * Dd * Hh * sizeof(__bf16);   // 8 MB
    int*   counts   = (int*)ws;     ws += 256;
    int*   list_tok = (int*)ws;     ws += (size_t)Ee * Tt * sizeof(int);           // 2 MB
    float* list_w   = (float*)ws;   /* + 2 MB */

    zero_init_kernel<<<4096, 256, 0, stream>>>(out, counts);
    convert_x_kernel<<<4096, 256, 0, stream>>>(token_stream, xb);
    convert_keys_kernel<<<2048, 256, 0, stream>>>(keys, keysT);
    convert_values_kernel<<<2048, 256, 0, stream>>>(values, valuesT);
    router_kernel<<<Tt / 8, 256, 0, stream>>>(selection_input, expert_sel, bias_ffn,
                                              sel_out, counts, list_tok, list_w);
    dim3 grid(Ee, Tt / MTILE);
    moe_ffn_kernel<<<grid, 256, 0, stream>>>(xb, keysT, valuesT,
                                             counts, list_tok, list_w, out);
}
